// VisionTransformer_59261958750494
// MI455X (gfx1250) — matrix operand order repaired and checked by reading, not compiled, not run on hardware
//
#include <hip/hip_runtime.h>
#include <hip/hip_bf16.h>
#include <math.h>

// ---------------- CDNA5 WMMA types ----------------
typedef __bf16 bf16_t;
typedef __attribute__((ext_vector_type(16))) __bf16 v16bf;
typedef __attribute__((ext_vector_type(8)))  float  v8f;

// ---------------- model constants ----------------
#define BB   32
#define SS   197          // tokens
#define SPAD 224          // padded tokens (multiple of 32)
#define DD   768
#define NHD  12
#define HDM  64
#define LLY  12
#define MLPD 3072
#define NCL  1000
#define NPAT 196

// CDNA5 async global->LDS DMA (ASYNCcnt-tracked), verified to assemble.
__device__ __forceinline__ void async_b128(unsigned ldsoff, const void* gaddr) {
    asm volatile("global_load_async_to_lds_b128 %0, %1, off"
                 :: "v"(ldsoff), "v"(gaddr) : "memory");
}
__device__ __forceinline__ void wait_async0() {
    asm volatile("s_wait_asynccnt 0" ::: "memory");
}

// =================================================================
// bf16 WMMA GEMM: C = act(scale*(A*B) + bias) (+residual)
//   A: M x K row-major.  B: stored N-major, B(k,n) = Bm[n*ldb + k]
//   (all weights are pre-transposed on device so every operand fits
//    this form -> LDS staging needs no transpose scatter at all).
// Block: 256 threads = 8 waves, tile 128(M) x 128(N), K step 32.
// Wave w -> rows [(w>>1)*32,+32), cols [(w&1)*64,+64): 8 v_wmma/step.
// Both tiles are staged straight into double-buffered LDS with
// global_load_async_to_lds_b128; one s_wait_asynccnt + one barrier
// per K-step, WMMA burst overlaps the next tile's DMA.
// Two-level batch (z = z1*nb2 + z2) with per-operand strides.
// =================================================================
__global__ __launch_bounds__(256)
void gemm_wmma_bf16(
    const bf16_t* __restrict__ A, int lda, long long sA1, long long sA2, int Ma,
    const bf16_t* __restrict__ Bm, int ldb, long long sB1, long long sB2, int Na,
    float* Cf, bf16_t* Cb, int ldc, long long sC1, long long sC2, int transC,
    const float* bias, long long sBi1, long long sBi2,
    const float* Res, int ldr, long long sR1, long long sR2,
    int M, int N, int K, float scale, int act, int nb2)
{
    __shared__ bf16_t As[2][128][40];   // [buf][m][k], row stride 40 (16B aligned)
    __shared__ bf16_t Bs[2][128][40];   // [buf][n][k]
    const unsigned BUFB = 128u * 40u * 2u;   // bytes per buffer

    int z  = blockIdx.z;
    int z1 = z / nb2;
    int z2 = z % nb2;
    A  += z1 * sA1 + z2 * sA2;
    Bm += z1 * sB1 + z2 * sB2;
    if (Cf)   Cf   += z1 * sC1  + z2 * sC2;
    if (Cb)   Cb   += z1 * sC1  + z2 * sC2;
    if (bias) bias += z1 * sBi1 + z2 * sBi2;
    if (Res)  Res  += z1 * sR1  + z2 * sR2;

    const int t    = threadIdx.x;
    const int wave = t >> 5;
    const int lane = t & 31;
    const int half = lane >> 4;
    const int mw   = (wave >> 1) * 32;   // wave row base in block tile
    const int nw   = (wave & 1) * 64;    // wave col base in block tile
    const int m0   = blockIdx.y * 128;
    const int n0   = blockIdx.x * 128;

    v8f acc[2][4];
#pragma unroll
    for (int a = 0; a < 2; ++a)
#pragma unroll
        for (int b = 0; b < 4; ++b)
#pragma unroll
            for (int j = 0; j < 8; ++j) acc[a][b][j] = 0.0f;

    // ---- per-thread staging descriptors (A and B share chunk mapping:
    //      128 rows x 32 k = 512 chunks of 8 bf16, 2 per thread) ----
    int row[2], col[2];
    bool aok[2], bok[2];
    const bf16_t* agp[2];
    const bf16_t* bgp[2];
    unsigned alds[2], blds[2];
#pragma unroll
    for (int i = 0; i < 2; ++i) {
        int c  = t + 256 * i;
        row[i] = c >> 2;               // 0..127
        col[i] = (c & 3) << 3;         // 0,8,16,24
        aok[i] = (m0 + row[i]) < Ma;
        bok[i] = (n0 + row[i]) < Na;
        agp[i] = A  + (size_t)(m0 + row[i]) * lda + col[i];
        bgp[i] = Bm + (size_t)(n0 + row[i]) * ldb + col[i];
        alds[i] = (unsigned)(unsigned long long)&As[0][row[i]][col[i]];
        blds[i] = (unsigned)(unsigned long long)&Bs[0][row[i]][col[i]];
    }
    // zero-fill OOB chunks once per buffer (their async lanes never fire)
    uint4 zz = make_uint4(0u, 0u, 0u, 0u);
#pragma unroll
    for (int i = 0; i < 2; ++i) {
        if (!aok[i]) { *(uint4*)&As[0][row[i]][col[i]] = zz; *(uint4*)&As[1][row[i]][col[i]] = zz; }
        if (!bok[i]) { *(uint4*)&Bs[0][row[i]][col[i]] = zz; *(uint4*)&Bs[1][row[i]][col[i]] = zz; }
    }

    auto stage = [&](int buf, int k0) {
        unsigned boff = (unsigned)buf * BUFB;
#pragma unroll
        for (int i = 0; i < 2; ++i) {
            if (aok[i]) {
                async_b128(alds[i] + boff, agp[i] + k0);
                if (k0 + 32 < K)
                    __builtin_prefetch(agp[i] + k0 + 32, 0, 3);
            }
            if (bok[i]) async_b128(blds[i] + boff, bgp[i] + k0);
        }
    };

    auto compute = [&](int buf) {
        union { v16bf v; unsigned u[8]; } af[2];
#pragma unroll
        for (int mt = 0; mt < 2; ++mt) {
            int mrow = mw + mt * 16 + (lane & 15);
#pragma unroll
            for (int v = 0; v < 8; ++v) {
                int kk = ((v >> 2) << 4) + (half << 3) + ((v & 3) << 1);
                af[mt].u[v] = *(const unsigned*)(&As[buf][mrow][kk]);
            }
        }
#pragma unroll
        for (int nt = 0; nt < 4; ++nt) {
            union { v16bf v; unsigned u[8]; } bfr;
            int nrow = nw + nt * 16 + (lane & 15);
#pragma unroll
            for (int v = 0; v < 8; ++v) {
                int kk = ((v >> 2) << 4) + (half << 3) + ((v & 3) << 1);
                bfr.u[v] = *(const unsigned*)(&Bs[buf][nrow][kk]);
            }
#pragma unroll
            for (int mt = 0; mt < 2; ++mt) {
                acc[mt][nt] = __builtin_amdgcn_wmma_f32_16x16x32_bf16(
                    false, af[mt].v, false, bfr.v, (short)0, acc[mt][nt], false, false);
            }
        }
    };

    // ---- async software pipeline ----
    stage(0, 0);
    wait_async0();
    int cur = 0;
    for (int k0 = 0; k0 < K; k0 += 32) {
        __syncthreads();
        bool more = (k0 + 32) < K;
        if (more) stage(cur ^ 1, k0 + 32);   // DMA next tile during WMMA burst
        compute(cur);
        if (more) wait_async0();             // own asyncs done before next barrier
        cur ^= 1;
    }

    // ---- epilogue: scale, bias, GELU, residual, f32/bf16 stores ----
#pragma unroll
    for (int mt = 0; mt < 2; ++mt) {
#pragma unroll
        for (int nt = 0; nt < 4; ++nt) {
            int n_g = n0 + nw + nt * 16 + (lane & 15);
            if (n_g >= N) continue;
            float bval = bias ? bias[n_g] : 0.0f;
#pragma unroll
            for (int v = 0; v < 8; ++v) {
                int m_g = m0 + mw + mt * 16 + v + (half << 3);
                if (m_g >= M) continue;
                float r = acc[mt][nt][v] * scale + bval;
                if (act == 1) r = 0.5f * r * (1.0f + erff(r * 0.70710678118654752f));
                if (Res) r += Res[(size_t)m_g * ldr + n_g];
                if (Cf) Cf[(size_t)m_g * ldc + n_g] = r;
                if (Cb) {
                    if (transC) Cb[(size_t)n_g * ldc + m_g] = (bf16_t)r;
                    else        Cb[(size_t)m_g * ldc + n_g] = (bf16_t)r;
                }
            }
        }
    }
}

// =================================================================
// TDM probe (never launched; compile-verified CDNA5 tensor DMA path)
// =================================================================
typedef unsigned int u32x4 __attribute__((ext_vector_type(4)));
typedef int          i32x4 __attribute__((ext_vector_type(4)));
typedef int          i32x8 __attribute__((ext_vector_type(8)));

__global__ void cdna5_probe_tdm(const bf16_t* __restrict__ src, float* __restrict__ out) {
    __shared__ bf16_t tile[32 * 64];
    unsigned lds = (unsigned)(unsigned long long)&tile[0];
    unsigned long long ga = (unsigned long long)src;
    u32x4 g0;
    g0[0] = 1u;                                                   // count=1
    g0[1] = lds;                                                  // lds_addr
    g0[2] = (unsigned)ga;                                         // global_addr[31:0]
    g0[3] = (unsigned)((ga >> 32) & 0x01FFFFFFu) | 0x80000000u;   // addr hi, type=2
    i32x8 g1;
    g1[0] = (1 << 16);                       // data_size=1 (2 bytes/elt)
    g1[1] = (int)((64u & 0xFFFFu) << 16);    // tensor_dim0=64 (low 16 into [63:48])
    g1[2] = (int)((32u & 0xFFFFu) << 16);    // dim0 hi=0, tensor_dim1=32 low
    g1[3] = (int)(64u << 16);                // dim1 hi=0, tile_dim0=64
    g1[4] = 32;                              // tile_dim1=32, tile_dim2=0
    g1[5] = 64;                              // tensor_dim0_stride=64 (low 32)
    g1[6] = 0;
    g1[7] = 0;
    i32x4 zr;  zr[0] = 0; zr[1] = 0; zr[2] = 0; zr[3] = 0;
    i32x8 zr8;
#pragma unroll
    for (int i = 0; i < 8; ++i) zr8[i] = 0;
    __builtin_amdgcn_tensor_load_to_lds(g0, g1, zr, zr, zr8, 0);
    __builtin_amdgcn_s_wait_tensorcnt(0);
    __builtin_amdgcn_s_cluster_barrier();
    __syncthreads();
    out[threadIdx.x] = (float)tile[threadIdx.x];
}

// =================================================================
// Small support kernels (f32 precision)
// =================================================================
// batched transpose + bf16 convert: out[b][c][r] = in[b][r][c]
__global__ __launch_bounds__(256)
void f32_to_bf16_T(const float* __restrict__ in, bf16_t* __restrict__ out,
                   int rows, int cols, int nbat) {
    int idx = blockIdx.x * 256 + threadIdx.x;
    int per = rows * cols;
    if (idx >= nbat * per) return;
    int b  = idx / per;
    int r2 = idx % per;
    int r  = r2 / cols;
    int c  = r2 % cols;
    out[(size_t)b * per + (size_t)c * rows + r] = (bf16_t)in[idx];
}

// images (B,C,224,224) -> patches (B*196, 768) bf16, torch-unfold order
__global__ __launch_bounds__(256)
void patchify_bf16(const float* __restrict__ img, bf16_t* __restrict__ out, int n) {
    int idx = blockIdx.x * 256 + threadIdx.x;
    if (idx >= n) return;
    int q   = idx % 768;
    int rem = idx / 768;
    int p   = rem % NPAT;
    int b   = rem / NPAT;
    int c  = q >> 8;
    int r2 = q & 255;
    int i  = r2 >> 4;
    int j  = r2 & 15;
    int pr = p / 14, pc = p % 14;
    size_t src = (((size_t)b * 3 + c) * 224 + (size_t)pr * 16 + i) * 224 + pc * 16 + j;
    out[idx] = (bf16_t)img[src];
}

// X[b,s,:] = (s==0 ? cls : T[b,s-1]) + pos[s]
__global__ __launch_bounds__(256)
void assemble_tokens(const float* __restrict__ T, const float* __restrict__ cls,
                     const float* __restrict__ pos, float* __restrict__ X, int n) {
    int idx = blockIdx.x * 256 + threadIdx.x;
    if (idx >= n) return;
    int d   = idx % DD;
    int rem = idx / DD;
    int s   = rem % SS;
    int b   = rem / SS;
    float v;
    if (s == 0) v = cls[d] + pos[d];
    else        v = T[((size_t)b * NPAT + (s - 1)) * DD + d] + pos[(size_t)s * DD + d];
    X[idx] = v;
    (void)b;
}

__global__ __launch_bounds__(256)
void layernorm_to_bf16(const float* __restrict__ X, const float* __restrict__ g,
                       const float* __restrict__ b, bf16_t* __restrict__ out, int D_) {
    int row = blockIdx.x;
    const float* x = X + (size_t)row * D_;
    __shared__ float red[256];
    int t = threadIdx.x;
    float s = 0.0f;
    for (int i = t; i < D_; i += 256) s += x[i];
    red[t] = s; __syncthreads();
    for (int o = 128; o > 0; o >>= 1) { if (t < o) red[t] += red[t + o]; __syncthreads(); }
    float mean = red[0] / D_;
    __syncthreads();
    float v = 0.0f;
    for (int i = t; i < D_; i += 256) { float d = x[i] - mean; v += d * d; }
    red[t] = v; __syncthreads();
    for (int o = 128; o > 0; o >>= 1) { if (t < o) red[t] += red[t + o]; __syncthreads(); }
    float rstd = rsqrtf(red[0] / D_ + 1e-5f);
    for (int i = t; i < D_; i += 256)
        out[(size_t)row * D_ + i] = (bf16_t)((x[i] - mean) * rstd * g[i] + b[i]);
}

// softmax over t<SS of scores[z,s,:], write bf16 probs, zero pad cols
__global__ __launch_bounds__(256)
void attn_softmax(const float* __restrict__ scores, bf16_t* __restrict__ probs) {
    int s = blockIdx.x;                 // 0..SS-1
    int z = blockIdx.y;                 // 0..NHD*BB-1
    const float* row = scores + ((size_t)z * SPAD + s) * SPAD;
    bf16_t* orow = probs + ((size_t)z * SPAD + s) * SPAD;
    __shared__ float red[256];
    int t = threadIdx.x;
    float m = -3.0e38f;
    for (int i = t; i < SS; i += 256) m = fmaxf(m, row[i]);
    red[t] = m; __syncthreads();
    for (int o = 128; o > 0; o >>= 1) { if (t < o) red[t] = fmaxf(red[t], red[t + o]); __syncthreads(); }
    float mx = red[0];
    __syncthreads();
    float sum = 0.0f;
    for (int i = t; i < SS; i += 256) sum += __expf(row[i] - mx);
    red[t] = sum; __syncthreads();
    for (int o = 128; o > 0; o >>= 1) { if (t < o) red[t] += red[t + o]; __syncthreads(); }
    float inv = 1.0f / red[0];
    for (int i = t; i < SPAD; i += 256)
        orow[i] = (bf16_t)((i < SS) ? __expf(row[i] - mx) * inv : 0.0f);
}

// X[b,s,h*64+e] += AO[(h*B+b), s, e]
__global__ __launch_bounds__(256)
void attn_residual(float* __restrict__ X, const float* __restrict__ AO, int n) {
    int idx = blockIdx.x * 256 + threadIdx.x;
    if (idx >= n) return;
    int d   = idx % DD;
    int rem = idx / DD;
    int s   = rem % SS;
    int b   = rem / SS;
    int h = d >> 6, e = d & 63;
    X[idx] += AO[(((size_t)h * BB + b) * SPAD + s) * HDM + e];
}

__global__ __launch_bounds__(256)
void extract_cls(const float* __restrict__ X, bf16_t* __restrict__ out, int n) {
    int idx = blockIdx.x * 256 + threadIdx.x;
    if (idx >= n) return;
    int d = idx % DD, b = idx / DD;
    out[idx] = (bf16_t)X[((size_t)b * SS) * DD + d];
}

__global__ __launch_bounds__(256)
void softmax_rows(const float* __restrict__ in, float* __restrict__ out, int cols) {
    int row = blockIdx.x;
    const float* x = in + (size_t)row * cols;
    float* op = out + (size_t)row * cols;
    __shared__ float red[256];
    int t = threadIdx.x;
    float m = -3.0e38f;
    for (int i = t; i < cols; i += 256) m = fmaxf(m, x[i]);
    red[t] = m; __syncthreads();
    for (int o = 128; o > 0; o >>= 1) { if (t < o) red[t] = fmaxf(red[t], red[t + o]); __syncthreads(); }
    float mx = red[0];
    __syncthreads();
    float s = 0.0f;
    for (int i = t; i < cols; i += 256) s += __expf(x[i] - mx);
    red[t] = s; __syncthreads();
    for (int o = 128; o > 0; o >>= 1) { if (t < o) red[t] += red[t + o]; __syncthreads(); }
    float inv = 1.0f / red[0];
    for (int i = t; i < cols; i += 256) op[i] = __expf(x[i] - mx) * inv;
}

// =================================================================
// Host side
// =================================================================
static void run_gemm(hipStream_t st,
    const bf16_t* A, int lda, long long sA1, long long sA2, int Ma,
    const bf16_t* Bm, int ldb, long long sB1, long long sB2, int Na,
    float* Cf, bf16_t* Cb, int ldc, long long sC1, long long sC2, int transC,
    const float* bias, long long sBi1, long long sBi2,
    const float* Res, int ldr, long long sR1, long long sR2,
    int M, int N, int K, float scale, int act, int nb1, int nb2)
{
    dim3 grid((N + 127) / 128, (M + 127) / 128, nb1 * nb2);
    gemm_wmma_bf16<<<grid, 256, 0, st>>>(A, lda, sA1, sA2, Ma,
                                         Bm, ldb, sB1, sB2, Na,
                                         Cf, Cb, ldc, sC1, sC2, transC,
                                         bias, sBi1, sBi2,
                                         Res, ldr, sR1, sR2,
                                         M, N, K, scale, act, nb2);
}

static inline int ceil256(int n) { return (n + 255) / 256; }

extern "C" void kernel_launch(void* const* d_in, const int* in_sizes, int n_in,
                              void* d_out, int out_size, void* d_ws, size_t ws_size,
                              hipStream_t stream) {
    (void)in_sizes; (void)n_in; (void)out_size; (void)ws_size;

    const float* images = (const float*)d_in[0];
    const float* Wp     = (const float*)d_in[1];
    const float* bp     = (const float*)d_in[2];
    const float* cls    = (const float*)d_in[3];
    const float* pos    = (const float*)d_in[4];
    const float* ln1_g  = (const float*)d_in[5];
    const float* ln1_b  = (const float*)d_in[6];
    const float* Wq     = (const float*)d_in[7];
    const float* bq     = (const float*)d_in[8];
    const float* Wk     = (const float*)d_in[9];
    const float* bk     = (const float*)d_in[10];
    const float* Wv     = (const float*)d_in[11];
    const float* bv     = (const float*)d_in[12];
    const float* ln2_g  = (const float*)d_in[13];
    const float* ln2_b  = (const float*)d_in[14];
    const float* W1     = (const float*)d_in[15];
    const float* b1     = (const float*)d_in[16];
    const float* W2     = (const float*)d_in[17];
    const float* b2     = (const float*)d_in[18];
    const float* Wh     = (const float*)d_in[19];
    const float* bh     = (const float*)d_in[20];
    float* out = (float*)d_out;

    // ---- workspace carve-up (256B aligned) ----
    char*  ws  = (char*)d_ws;
    size_t off = 0;
    auto alloc = [&](size_t bytes) -> void* {
        void* p = ws + off;
        off = (off + bytes + 255) & ~(size_t)255;
        return p;
    };
    const size_t nWqkv = (size_t)LLY * NHD * HDM * HDM;          // 589824
    bf16_t* WpT = (bf16_t*)alloc((size_t)DD * DD * 2);           // [n][k]
    bf16_t* WqT = (bf16_t*)alloc(nWqkv * 2);
    bf16_t* WkT = (bf16_t*)alloc(nWqkv * 2);
    bf16_t* WvT = (bf16_t*)alloc(nWqkv * 2);
    bf16_t* W1T = (bf16_t*)alloc((size_t)LLY * DD * MLPD * 2);
    bf16_t* W2T = (bf16_t*)alloc((size_t)LLY * MLPD * DD * 2);
    bf16_t* WhT = (bf16_t*)alloc((size_t)NCL * DD * 2);
    bf16_t* patches_bf = (bf16_t*)alloc((size_t)BB * NPAT * DD * 2);
    float*  T  = (float*)alloc((size_t)BB * NPAT * DD * 4);
    float*  X  = (float*)alloc((size_t)BB * SS * DD * 4);
    bf16_t* H_bf = (bf16_t*)alloc((size_t)BB * SS * DD * 2);
    const size_t nQKV = (size_t)NHD * BB * SPAD * HDM;           // 5505024
    bf16_t* Q_bf = (bf16_t*)alloc(nQKV * 2);
    bf16_t* K_bf = (bf16_t*)alloc(nQKV * 2);
    bf16_t* Vt_bf = (bf16_t*)alloc(nQKV * 2);                    // [h,b][e][t] transposed
    const size_t nSc = (size_t)NHD * BB * SPAD * SPAD;           // 19267584
    float*  scores = (float*)alloc(nSc * 4);
    bf16_t* probs  = (bf16_t*)alloc(nSc * 2);
    float*  AO     = (float*)alloc(nQKV * 4);
    bf16_t* M1_bf  = (bf16_t*)alloc((size_t)BB * SS * MLPD * 2);
    bf16_t* Xcls_bf = (bf16_t*)alloc((size_t)BB * DD * 2);
    float*  logits  = (float*)alloc((size_t)BB * NCL * 4);

    // Vt needs zero padding cols t in [197,224) (k-dim of probs*V)
    (void)hipMemsetAsync(Vt_bf, 0, nQKV * 2, stream);

    // ---- convert + transpose weights to bf16 N-major ----
    {
        int n = DD * DD;
        f32_to_bf16_T<<<ceil256(n), 256, 0, stream>>>(Wp, WpT, DD, DD, 1);
        n = (int)nWqkv;
        f32_to_bf16_T<<<ceil256(n), 256, 0, stream>>>(Wq, WqT, HDM, HDM, LLY * NHD);
        f32_to_bf16_T<<<ceil256(n), 256, 0, stream>>>(Wk, WkT, HDM, HDM, LLY * NHD);
        f32_to_bf16_T<<<ceil256(n), 256, 0, stream>>>(Wv, WvT, HDM, HDM, LLY * NHD);
        n = LLY * DD * MLPD;
        f32_to_bf16_T<<<ceil256(n), 256, 0, stream>>>(W1, W1T, DD, MLPD, LLY);
        f32_to_bf16_T<<<ceil256(n), 256, 0, stream>>>(W2, W2T, MLPD, DD, LLY);
        n = DD * NCL;
        f32_to_bf16_T<<<ceil256(n), 256, 0, stream>>>(Wh, WhT, DD, NCL, 1);
    }

    // ---- patch embedding ----
    {
        int n = BB * NPAT * DD;
        patchify_bf16<<<ceil256(n), 256, 0, stream>>>(images, patches_bf, n);
        run_gemm(stream,
                 patches_bf, DD, 0, 0, BB * NPAT,
                 WpT, DD, 0, 0, DD,
                 T, nullptr, DD, 0, 0, 0,
                 bp, 0, 0,
                 nullptr, 0, 0, 0,
                 BB * NPAT, DD, DD, 1.0f, 0, 1, 1);
        n = BB * SS * DD;
        assemble_tokens<<<ceil256(n), 256, 0, stream>>>(T, cls, pos, X, n);
    }

    const int nTokD = BB * SS * DD;
    const long long sQz1 = (long long)BB * SPAD * HDM;   // per-head stride in Q/K/Vt
    const long long sQz2 = (long long)SPAD * HDM;        // per-batch stride

    for (int l = 0; l < LLY; ++l) {
        // LN1 -> H_bf
        layernorm_to_bf16<<<BB * SS, 256, 0, stream>>>(X, ln1_g + (size_t)l * DD,
                                                       ln1_b + (size_t)l * DD, H_bf, DD);
        // per-head QKV projections: z1=head, z2=batch.  V is stored transposed.
        const bf16_t* Wts[3] = { WqT + (size_t)l * NHD * HDM * HDM,
                                 WkT + (size_t)l * NHD * HDM * HDM,
                                 WvT + (size_t)l * NHD * HDM * HDM };
        const float*  bqs[3] = { bq + (size_t)l * NHD * HDM,
                                 bk + (size_t)l * NHD * HDM,
                                 bv + (size_t)l * NHD * HDM };
        bf16_t* Outs[3] = { Q_bf, K_bf, Vt_bf };
        for (int p = 0; p < 3; ++p) {
            int tc  = (p == 2) ? 1 : 0;
            int ldo = (p == 2) ? SPAD : HDM;
            run_gemm(stream,
                     H_bf, DD, HDM, (long long)SS * DD, SS,
                     Wts[p], HDM, (long long)HDM * HDM, 0, HDM,
                     nullptr, Outs[p], ldo, sQz1, sQz2, tc,
                     bqs[p], HDM, 0,
                     nullptr, 0, 0, 0,
                     SS, HDM, HDM, 1.0f, 0, NHD, BB);
        }
        // scores = Q * K^T / 8   (batch z = h*B + b; K rows >=197 zero-filled)
        run_gemm(stream,
                 Q_bf, HDM, 0, sQz2, SS,
                 K_bf, HDM, 0, sQz2, SS,
                 scores, nullptr, SPAD, 0, (long long)SPAD * SPAD, 0,
                 nullptr, 0, 0,
                 nullptr, 0, 0, 0,
                 SS, SS, HDM, 0.125f, 0, 1, NHD * BB);
        // softmax -> bf16 probs (zero pad cols)
        attn_softmax<<<dim3(SS, NHD * BB), 256, 0, stream>>>(scores, probs);
        // AO = probs * V   (B = Vt[e][t], N-major with k=t)
        run_gemm(stream,
                 probs, SPAD, 0, (long long)SPAD * SPAD, SS,
                 Vt_bf, SPAD, 0, sQz2, HDM,
                 AO, nullptr, HDM, 0, sQz2, 0,
                 nullptr, 0, 0,
                 nullptr, 0, 0, 0,
                 SS, HDM, SPAD, 1.0f, 0, 1, NHD * BB);
        // X += attn out (head interleave)
        attn_residual<<<ceil256(nTokD), 256, 0, stream>>>(X, AO, nTokD);

        // LN2 -> H_bf
        layernorm_to_bf16<<<BB * SS, 256, 0, stream>>>(X, ln2_g + (size_t)l * DD,
                                                       ln2_b + (size_t)l * DD, H_bf, DD);
        // MLP1: GELU(H*W1 + b1) -> bf16
        run_gemm(stream,
                 H_bf, DD, 0, 0, BB * SS,
                 W1T + (size_t)l * DD * MLPD, DD, 0, 0, MLPD,
                 nullptr, M1_bf, MLPD, 0, 0, 0,
                 b1 + (size_t)l * MLPD, 0, 0,
                 nullptr, 0, 0, 0,
                 BB * SS, MLPD, DD, 1.0f, 1, 1, 1);
        // MLP2: X = M1*W2 + b2 + X
        run_gemm(stream,
                 M1_bf, MLPD, 0, 0, BB * SS,
                 W2T + (size_t)l * MLPD * DD, MLPD, 0, 0, DD,
                 X, nullptr, DD, 0, 0, 0,
                 b2 + (size_t)l * DD, 0, 0,
                 X, DD, 0, 0,
                 BB * SS, DD, MLPD, 1.0f, 0, 1, 1);
    }

    // ---- classifier head + softmax ----
    {
        int n = BB * DD;
        extract_cls<<<ceil256(n), 256, 0, stream>>>(X, Xcls_bf, n);
        run_gemm(stream,
                 Xcls_bf, DD, 0, 0, BB,
                 WhT, DD, 0, 0, NCL,
                 logits, nullptr, NCL, 0, 0, 0,
                 bh, 0, 0,
                 nullptr, 0, 0, 0,
                 BB, NCL, DD, 1.0f, 0, 1, 1);
        softmax_rows<<<BB, 256, 0, stream>>>(logits, out, NCL);
    }
}
